// MixtralMoE_49185965473905
// MI455X (gfx1250) — matrix operand order repaired and checked by reading, not compiled, not run on hardware
//
#include <hip/hip_runtime.h>

#define NUM_EXPERTS 8
#define TOP_K 2
#define HID 2048
#define FFN 7168
#define TOK 4096

typedef __attribute__((ext_vector_type(8)))  unsigned short v8u16;
typedef __attribute__((ext_vector_type(16))) unsigned short v16u16;
typedef __attribute__((ext_vector_type(16))) __bf16         v16bf;
typedef __attribute__((ext_vector_type(8)))  float          v8f;

__device__ __forceinline__ unsigned short f32_to_bf16(float f) {
  union { float f; unsigned int u; } c; c.f = f;
  unsigned int u = c.u;
  unsigned int r = u + 0x7FFFu + ((u >> 16) & 1u);      // round-to-nearest-even
  if ((u & 0x7F800000u) == 0x7F800000u) r = u;          // preserve Inf/NaN
  return (unsigned short)(r >> 16);
}

// Assemble a v16bf WMMA fragment from two 16-byte LDS chunks (low 8, high 8).
__device__ __forceinline__ v16bf make_frag(const unsigned short* lds, int off_lo, int off_hi) {
  v8u16 lo = *(const v8u16*)(lds + off_lo);
  v8u16 hi = *(const v8u16*)(lds + off_hi);
  v16u16 c = __builtin_shufflevector(lo, hi, 0,1,2,3,4,5,6,7,8,9,10,11,12,13,14,15);
  return __builtin_bit_cast(v16bf, c);
}

// CDNA5 async global->LDS copy (16B per lane), tracked by ASYNCcnt.
__device__ __forceinline__ void async_b128(unsigned lds_off, const void* gaddr) {
  asm volatile("global_load_async_to_lds_b128 %0, %1, off"
               :: "v"(lds_off), "v"(gaddr) : "memory");
}
__device__ __forceinline__ unsigned lds_off_u32(const void* p) {
  return (unsigned)(uintptr_t)p;     // low 32 bits of flat LDS address = LDS offset
}

// ---------------------------------------------------------------- utilities
__global__ void zero_counts_kernel(int* counts) {
  if (threadIdx.x < NUM_EXPERTS) counts[threadIdx.x] = 0;
}

// -------------------------------------------------- weight staging (per expert)
// src[K][N] fp32  ->  dst[N][K] bf16 (transposed, so GEMM B-tiles are K-contiguous)
__global__ __launch_bounds__(256) void transpose_cvt_kernel(
    const float* __restrict__ src, unsigned short* __restrict__ dst, int K, int N)
{
  __shared__ unsigned short t[64 * 72];
  const int k0 = blockIdx.y * 64, n0 = blockIdx.x * 64;
  const int tid = threadIdx.x;
#pragma unroll
  for (int i = 0; i < 16; ++i) {
    int idx = tid + i * 256;                 // 0..4095
    int r = idx >> 6, c = idx & 63;
    t[r * 72 + c] = f32_to_bf16(src[(size_t)(k0 + r) * N + n0 + c]);
  }
  __syncthreads();
#pragma unroll
  for (int i = 0; i < 16; ++i) {
    int idx = tid + i * 256;
    int n = idx >> 6, kk = idx & 63;
    dst[(size_t)(n0 + n) * K + k0 + kk] = t[kk * 72 + n];
  }
}

// ------------------------------------------------------------------- router
__global__ __launch_bounds__(256) void router_kernel(
    const float* __restrict__ x, const float* __restrict__ gw,
    unsigned short* __restrict__ Xb, float* __restrict__ token_w,
    int* __restrict__ counts, int* __restrict__ idxlist)
{
  __shared__ float red[256 * NUM_EXPERTS];
  const int t = blockIdx.x, tid = threadIdx.x;
  float part[NUM_EXPERTS];
#pragma unroll
  for (int e = 0; e < NUM_EXPERTS; ++e) part[e] = 0.f;

  for (int i = tid; i < HID; i += 256) {
    float xv = x[(size_t)t * HID + i];
    Xb[(size_t)t * HID + i] = f32_to_bf16(xv);
#pragma unroll
    for (int e = 0; e < NUM_EXPERTS; ++e) part[e] += xv * gw[i * NUM_EXPERTS + e];
  }
#pragma unroll
  for (int e = 0; e < NUM_EXPERTS; ++e) red[tid * NUM_EXPERTS + e] = part[e];
  __syncthreads();
  for (int s = 128; s > 0; s >>= 1) {
    if (tid < s) {
#pragma unroll
      for (int e = 0; e < NUM_EXPERTS; ++e)
        red[tid * NUM_EXPERTS + e] += red[(tid + s) * NUM_EXPERTS + e];
    }
    __syncthreads();
  }
  if (tid == 0) {
    float lg[NUM_EXPERTS];
#pragma unroll
    for (int e = 0; e < NUM_EXPERTS; ++e) lg[e] = red[e];
    float mx = lg[0];
#pragma unroll
    for (int e = 1; e < NUM_EXPERTS; ++e) mx = fmaxf(mx, lg[e]);
    float ex[NUM_EXPERTS], sum = 0.f;
#pragma unroll
    for (int e = 0; e < NUM_EXPERTS; ++e) { ex[e] = __expf(lg[e] - mx); sum += ex[e]; }
    int i0 = 0;
#pragma unroll
    for (int e = 1; e < NUM_EXPERTS; ++e) if (ex[e] > ex[i0]) i0 = e;
    int i1 = (i0 == 0) ? 1 : 0;
#pragma unroll
    for (int e = 0; e < NUM_EXPERTS; ++e) if (e != i0 && ex[e] > ex[i1]) i1 = e;
    float p0 = ex[i0] / sum, p1 = ex[i1] / sum, wsum = p0 + p1;
    token_w[t * 2 + 0] = p0 / wsum;
    token_w[t * 2 + 1] = p1 / wsum;
    int pos0 = atomicAdd(&counts[i0], 1);
    idxlist[i0 * TOK + pos0] = (t << 1);
    int pos1 = atomicAdd(&counts[i1], 1);
    idxlist[i1 * TOK + pos1] = (t << 1) | 1;
  }
}

// ------------------------------------------------- expert FFN stage 1 (W1/W3)
// Block tile 128(M) x 128(N); 8 waves in 4x2 grid, each 32x64 => 16 bf16 WMMA
// accumulators (W1+W3 fused). Async double-buffered LDS pipeline.
__global__ __launch_bounds__(256) void ffn1_kernel(
    const unsigned short* __restrict__ Xb,
    const unsigned short* __restrict__ w1b,   // [FFN][HID] bf16 (transposed)
    const unsigned short* __restrict__ w3b,
    const int* __restrict__ idxlist, const int* __restrict__ cnt,
    unsigned short* __restrict__ Hbuf)
{
  __shared__ __align__(16) unsigned short ldsA [2][128 * 32];
  __shared__ __align__(16) unsigned short ldsB1[2][128 * 32];
  __shared__ __align__(16) unsigned short ldsB3[2][128 * 32];
  __shared__ int rowTok[128];

  const int count = cnt[0];
  const int m0 = blockIdx.y * 128;
  if (m0 >= count) return;
  const int n0 = blockIdx.x * 128;
  const int tid = threadIdx.x;
  const int lane = tid & 31, wv = tid >> 5;
  const int waveM = wv >> 1, waveN = wv & 1;          // 4 x 2 wave grid
  const int halfsel = lane >> 4, lrow = lane & 15;

  if (tid < 128) {
    int p = m0 + tid;
    rowTok[tid] = (p < count) ? (idxlist[p] >> 1) : 0;
  }
  __syncthreads();

  // Loader mappings: A 128x32 = 512 x 16B (2 per thread); B 128x32 (1 per thread)
  const int ar = tid >> 2, ag = (tid & 3) * 8;
  const int bn = tid >> 1, bh = (tid & 1) * 16;
  const unsigned short* aSrc0 = Xb  + (size_t)rowTok[ar]      * HID + ag;
  const unsigned short* aSrc1 = Xb  + (size_t)rowTok[ar + 64] * HID + ag;
  const unsigned short* b1Src = w1b + (size_t)(n0 + bn) * HID + bh;
  const unsigned short* b3Src = w3b + (size_t)(n0 + bn) * HID + bh;
  const unsigned aOff0 = ar * 32 + ag;
  const unsigned aOff1 = (ar + 64) * 32 + ag;
  const unsigned bOff  = bn * 32 + bh;

  v8f acc1[2][4], acc3[2][4];
#pragma unroll
  for (int mi = 0; mi < 2; ++mi)
#pragma unroll
    for (int ni = 0; ni < 4; ++ni) {
      acc1[mi][ni] = (v8f){0.f,0.f,0.f,0.f,0.f,0.f,0.f,0.f};
      acc3[mi][ni] = (v8f){0.f,0.f,0.f,0.f,0.f,0.f,0.f,0.f};
    }

  // Prologue: issue tile 0 into buffer 0.
  async_b128(lds_off_u32(&ldsA [0][aOff0]), aSrc0);
  async_b128(lds_off_u32(&ldsA [0][aOff1]), aSrc1);
  async_b128(lds_off_u32(&ldsB1[0][bOff]),  b1Src);
  async_b128(lds_off_u32(&ldsB3[0][bOff]),  b3Src);

  const int nIter = HID / 32;
  for (int it = 0; it < nIter; ++it) {
    const int buf = it & 1;
    const bool more = (it + 1) < nIter;
    if (more) {
      const int k1 = (it + 1) * 32, nb = buf ^ 1;
      async_b128(lds_off_u32(&ldsA [nb][aOff0]), aSrc0 + k1);
      async_b128(lds_off_u32(&ldsA [nb][aOff1]), aSrc1 + k1);
      async_b128(lds_off_u32(&ldsB1[nb][bOff]),  b1Src + k1);
      async_b128(lds_off_u32(&ldsB3[nb][bOff]),  b3Src + k1);
      asm volatile("s_wait_asynccnt 4" ::: "memory");   // tile `it` landed (in-order)
    } else {
      asm volatile("s_wait_asynccnt 0" ::: "memory");
    }
    __syncthreads();

    const unsigned short* A  = ldsA [buf];
    const unsigned short* B1 = ldsB1[buf];
    const unsigned short* B3 = ldsB3[buf];
    v16bf aF[2];
#pragma unroll
    for (int mi = 0; mi < 2; ++mi) {
      int row = waveM * 32 + mi * 16 + lrow;
      aF[mi] = make_frag(A, row * 32 + halfsel * 16, row * 32 + halfsel * 16 + 8);
    }
#pragma unroll
    for (int ni = 0; ni < 4; ++ni) {
      int col = waveN * 64 + ni * 16 + lrow;
      v16bf b1 = make_frag(B1, col * 32 + halfsel * 16, col * 32 + halfsel * 16 + 8);
      v16bf b3 = make_frag(B3, col * 32 + halfsel * 16, col * 32 + halfsel * 16 + 8);
#pragma unroll
      for (int mi = 0; mi < 2; ++mi) {
        acc1[mi][ni] = __builtin_amdgcn_wmma_f32_16x16x32_bf16(
            false, aF[mi], false, b1, (short)0, acc1[mi][ni], false, false);
        acc3[mi][ni] = __builtin_amdgcn_wmma_f32_16x16x32_bf16(
            false, aF[mi], false, b3, (short)0, acc3[mi][ni], false, false);
      }
    }
    __syncthreads();   // protect buf before it is overwritten next iteration
  }

  // Epilogue: h = silu(y1) * y3 -> bf16 H buffer (rows = position in expert list)
#pragma unroll
  for (int mi = 0; mi < 2; ++mi)
#pragma unroll
    for (int ni = 0; ni < 4; ++ni) {
      int colBase = n0 + waveN * 64 + ni * 16 + lrow;
#pragma unroll
      for (int j = 0; j < 8; ++j) {
        int rowT = waveM * 32 + mi * 16 + halfsel * 8 + j;
        int pos = m0 + rowT;
        if (pos < count) {
          float y1 = acc1[mi][ni][j];
          float y3 = acc3[mi][ni][j];
          float s = y1 / (1.0f + __expf(-y1));
          Hbuf[(size_t)pos * FFN + colBase] = f32_to_bf16(s * y3);
        }
      }
    }
}

// ------------------------------------------------- expert FFN stage 2 (W2)
__global__ __launch_bounds__(256) void ffn2_kernel(
    const unsigned short* __restrict__ Hbuf,
    const unsigned short* __restrict__ w2b,   // [HID][FFN] bf16 (transposed)
    const int* __restrict__ idxlist, const int* __restrict__ cnt,
    const float* __restrict__ token_w,
    float* __restrict__ partial)
{
  __shared__ __align__(16) unsigned short ldsA[2][128 * 32];
  __shared__ __align__(16) unsigned short ldsB[2][128 * 32];
  __shared__ int rowEnt[128];
  __shared__ float rowW[128];

  const int count = cnt[0];
  const int m0 = blockIdx.y * 128;
  if (m0 >= count) return;
  const int n0 = blockIdx.x * 128;
  const int tid = threadIdx.x;
  const int lane = tid & 31, wv = tid >> 5;
  const int waveM = wv >> 1, waveN = wv & 1;
  const int halfsel = lane >> 4, lrow = lane & 15;

  if (tid < 128) {
    int p = m0 + tid;
    int ent = (p < count) ? idxlist[p] : 0;
    rowEnt[tid] = ent;
    rowW[tid] = (p < count) ? token_w[ent] : 0.f;
  }
  __syncthreads();

  const int ar = tid >> 2, ag = (tid & 3) * 8;
  const int bn = tid >> 1, bh = (tid & 1) * 16;
  const unsigned short* aSrc0 = Hbuf + (size_t)(m0 + ar)      * FFN + ag;
  const unsigned short* aSrc1 = Hbuf + (size_t)(m0 + ar + 64) * FFN + ag;
  const unsigned short* bSrc  = w2b  + (size_t)(n0 + bn) * FFN + bh;
  const unsigned aOff0 = ar * 32 + ag;
  const unsigned aOff1 = (ar + 64) * 32 + ag;
  const unsigned bOff  = bn * 32 + bh;

  v8f acc[2][4];
#pragma unroll
  for (int mi = 0; mi < 2; ++mi)
#pragma unroll
    for (int ni = 0; ni < 4; ++ni)
      acc[mi][ni] = (v8f){0.f,0.f,0.f,0.f,0.f,0.f,0.f,0.f};

  async_b128(lds_off_u32(&ldsA[0][aOff0]), aSrc0);
  async_b128(lds_off_u32(&ldsA[0][aOff1]), aSrc1);
  async_b128(lds_off_u32(&ldsB[0][bOff]),  bSrc);

  const int nIter = FFN / 32;
  for (int it = 0; it < nIter; ++it) {
    const int buf = it & 1;
    const bool more = (it + 1) < nIter;
    if (more) {
      const int k1 = (it + 1) * 32, nb = buf ^ 1;
      async_b128(lds_off_u32(&ldsA[nb][aOff0]), aSrc0 + k1);
      async_b128(lds_off_u32(&ldsA[nb][aOff1]), aSrc1 + k1);
      async_b128(lds_off_u32(&ldsB[nb][bOff]),  bSrc + k1);
      asm volatile("s_wait_asynccnt 3" ::: "memory");
    } else {
      asm volatile("s_wait_asynccnt 0" ::: "memory");
    }
    __syncthreads();

    const unsigned short* A = ldsA[buf];
    const unsigned short* B = ldsB[buf];
    v16bf aF[2];
#pragma unroll
    for (int mi = 0; mi < 2; ++mi) {
      int row = waveM * 32 + mi * 16 + lrow;
      aF[mi] = make_frag(A, row * 32 + halfsel * 16, row * 32 + halfsel * 16 + 8);
    }
#pragma unroll
    for (int ni = 0; ni < 4; ++ni) {
      int col = waveN * 64 + ni * 16 + lrow;
      v16bf bF = make_frag(B, col * 32 + halfsel * 16, col * 32 + halfsel * 16 + 8);
#pragma unroll
      for (int mi = 0; mi < 2; ++mi)
        acc[mi][ni] = __builtin_amdgcn_wmma_f32_16x16x32_bf16(
            false, aF[mi], false, bF, (short)0, acc[mi][ni], false, false);
    }
    __syncthreads();
  }

#pragma unroll
  for (int mi = 0; mi < 2; ++mi)
#pragma unroll
    for (int ni = 0; ni < 4; ++ni) {
      int colBase = n0 + waveN * 64 + ni * 16 + lrow;
#pragma unroll
      for (int j = 0; j < 8; ++j) {
        int rowT = waveM * 32 + mi * 16 + halfsel * 8 + j;
        int pos = m0 + rowT;
        if (pos < count) {
          int ent = rowEnt[rowT];
          partial[(size_t)ent * HID + colBase] = acc[mi][ni][j] * rowW[rowT];
        }
      }
    }
}

// --------------------------------------------------------------- final sum
__global__ __launch_bounds__(256) void final_kernel(
    const float* __restrict__ partial, float* __restrict__ out)
{
  size_t i = (size_t)blockIdx.x * 256 + threadIdx.x;   // i = t*HID + c
  size_t t = i >> 11;
  size_t c = i & (HID - 1);
  out[i] = partial[(t * 2) * HID + c] + partial[(t * 2 + 1) * HID + c];
}

// ------------------------------------------------------------------- launch
extern "C" void kernel_launch(void* const* d_in, const int* in_sizes, int n_in,
                              void* d_out, int out_size, void* d_ws, size_t ws_size,
                              hipStream_t stream) {
  const float* x   = (const float*)d_in[0];
  const float* gw  = (const float*)d_in[1];
  const float* w1s = (const float*)d_in[2];
  const float* w3s = (const float*)d_in[3];
  const float* w2s = (const float*)d_in[4];
  float* out = (float*)d_out;

  char* ws = (char*)d_ws;
  size_t off = 0;
  auto alloc = [&](size_t bytes) -> void* {
    void* p = ws + off;
    off = (off + bytes + 255) & ~(size_t)255;
    return p;
  };
  unsigned short* Xb      = (unsigned short*)alloc((size_t)TOK * HID * 2);
  unsigned short* Hbuf    = (unsigned short*)alloc((size_t)TOK * FFN * 2);
  float*          partial = (float*)alloc((size_t)TOK * 2 * HID * 4);
  float*          token_w = (float*)alloc((size_t)TOK * 2 * 4);
  int*            counts  = (int*)alloc(NUM_EXPERTS * 4);
  int*            idxlist = (int*)alloc((size_t)NUM_EXPERTS * TOK * 4);
  unsigned short* w1b     = (unsigned short*)alloc((size_t)FFN * HID * 2);
  unsigned short* w3b     = (unsigned short*)alloc((size_t)FFN * HID * 2);
  unsigned short* w2b     = (unsigned short*)alloc((size_t)HID * FFN * 2);

  zero_counts_kernel<<<1, 32, 0, stream>>>(counts);
  router_kernel<<<TOK, 256, 0, stream>>>(x, gw, Xb, token_w, counts, idxlist);

  for (int e = 0; e < NUM_EXPERTS; ++e) {
    const float* w1 = w1s + (size_t)e * HID * FFN;
    const float* w3 = w3s + (size_t)e * HID * FFN;
    const float* w2 = w2s + (size_t)e * FFN * HID;
    // Stage expert weights: fp32 [K][N] -> bf16 [N][K] (fits in L2 for reuse)
    transpose_cvt_kernel<<<dim3(FFN / 64, HID / 64), 256, 0, stream>>>(w1, w1b, HID, FFN);
    transpose_cvt_kernel<<<dim3(FFN / 64, HID / 64), 256, 0, stream>>>(w3, w3b, HID, FFN);
    transpose_cvt_kernel<<<dim3(HID / 64, FFN / 64), 256, 0, stream>>>(w2, w2b, FFN, HID);

    ffn1_kernel<<<dim3(FFN / 128, TOK / 128), 256, 0, stream>>>(
        Xb, w1b, w3b, idxlist + e * TOK, counts + e, Hbuf);
    ffn2_kernel<<<dim3(HID / 128, TOK / 128), 256, 0, stream>>>(
        Hbuf, w2b, idxlist + e * TOK, counts + e, token_w, partial);
  }
  final_kernel<<<(TOK * HID) / 256, 256, 0, stream>>>(partial, out);
}
